// MoE_Layer_28527172780757
// MI455X (gfx1250) — matrix operand order repaired and checked by reading, not compiled, not run on hardware
//
#include <hip/hip_runtime.h>

#define N_TOK 16384
#define DIM   1024
#define HID   128
#define OUTD  1024
#define NEXP  16

typedef __attribute__((ext_vector_type(16))) __bf16 v16bf;
typedef __attribute__((ext_vector_type(8)))  float  v8f;
typedef __attribute__((ext_vector_type(4)))  unsigned int u32x4;

__device__ __forceinline__ unsigned short f2bf(float f) {
    unsigned int u = __float_as_uint(f);
    u += 0x7FFFu + ((u >> 16) & 1u);          // RTNE
    return (unsigned short)(u >> 16);
}

__device__ __forceinline__ v8f zero8() {
    v8f z = {0.f, 0.f, 0.f, 0.f, 0.f, 0.f, 0.f, 0.f};
    return z;
}

// Fragment region layout: [half(2)][lane(32)][8 bf16] = 512 u16 = 1 KB
__device__ __forceinline__ v16bf load_frag(const unsigned short* region, int lane) {
    v16bf f;
    ((u32x4*)&f)[0] = *(const u32x4*)(region + lane * 8);
    ((u32x4*)&f)[1] = *(const u32x4*)(region + 256 + lane * 8);
    return f;
}

// Async-copy one 16 KB chunk (1024 x b128) global -> LDS, 128 threads, ASYNCcnt-tracked.
__device__ __forceinline__ void async_copy_chunk(const unsigned short* gbase,
                                                 unsigned int lds_base, int tid) {
    unsigned long long sbase = (unsigned long long)(size_t)gbase;
#pragma unroll
    for (int q = 0; q < 8; ++q) {
        unsigned int voff = (unsigned int)(q * 128 + tid) * 16u;
        unsigned int ldst = lds_base + voff;
        asm volatile("global_load_async_to_lds_b128 %0, %1, %2"
                     :: "v"(ldst), "v"(voff), "s"(sbase) : "memory");
    }
}

__device__ __forceinline__ void wait_async0() {
    asm volatile("s_wait_asynccnt 0x0" ::: "memory");
}

// ---------------- K0: zero output ----------------
__global__ void zero_kernel(float* __restrict__ out, long long n) {
    long long i = (long long)blockIdx.x * blockDim.x + threadIdx.x;
    long long stride = (long long)gridDim.x * blockDim.x;
    for (; i < n; i += stride) out[i] = 0.0f;
}

// ---------------- K1: gating (logits, top2, softmax partials) ----------------
__global__ __launch_bounds__(256) void gate_kernel(
        const float* __restrict__ x, const float* __restrict__ wg,
        int* __restrict__ topIdx, float* __restrict__ slotW,
        float* __restrict__ gpart) {
    __shared__ float xs[64][65];
    __shared__ float wsc[64][16];
    __shared__ float lg[64][16];
    int tid = threadIdx.x;
    int tok = tid & 63;
    int eg  = tid >> 6;
    int tokBase = blockIdx.x * 64;
    float acc[4] = {0.f, 0.f, 0.f, 0.f};

    for (int kc = 0; kc < DIM; kc += 64) {
        for (int i = tid; i < 64 * 64; i += 256) {
            int r = i >> 6, c = i & 63;
            xs[r][c] = x[(size_t)(tokBase + r) * DIM + kc + c];
        }
        for (int i = tid; i < 64 * 16; i += 256) {
            int d = i >> 4, ee = i & 15;
            wsc[d][ee] = wg[(size_t)(kc + d) * NEXP + ee];
        }
        __syncthreads();
        for (int d = 0; d < 64; ++d) {
            float xv = xs[tok][d];
#pragma unroll
            for (int j = 0; j < 4; ++j) acc[j] += xv * wsc[d][eg * 4 + j];
        }
        __syncthreads();
    }
#pragma unroll
    for (int j = 0; j < 4; ++j) lg[tok][eg * 4 + j] = acc[j];
    __syncthreads();

    if (tid < 64) {
        float l[16];
#pragma unroll
        for (int e = 0; e < 16; ++e) l[e] = lg[tid][e];
        int i1 = 0; float m1 = l[0];
        for (int e = 1; e < 16; ++e) if (l[e] > m1) { m1 = l[e]; i1 = e; }
        int i2 = (i1 == 0) ? 1 : 0; float m2 = l[i2];
        for (int e = 0; e < 16; ++e) {
            if (e == i1 || e == i2) continue;
            if (l[e] > m2) { m2 = l[e]; i2 = e; }
        }
        float t  = expf(m2 - m1);
        float wA = 1.0f / (1.0f + t);
        float wB = t / (1.0f + t);
        int g = tokBase + tid;
        topIdx[g * 2 + 0] = i1; topIdx[g * 2 + 1] = i2;
        slotW[g * 2 + 0]  = wA; slotW[g * 2 + 1]  = wB;
        float s = 0.f;
#pragma unroll
        for (int e = 0; e < 16; ++e) { float v = expf(l[e] - m1); l[e] = v; s += v; }
        float inv = 1.0f / s;
#pragma unroll
        for (int e = 0; e < 16; ++e) lg[tid][e] = l[e] * inv;
    }
    __syncthreads();
    if (tid < 16) {
        float s = 0.f;
        for (int r = 0; r < 64; ++r) s += lg[r][tid];   // fixed order: deterministic
        gpart[blockIdx.x * 16 + tid] = s;
    }
}

// ---------------- K2: per-chunk expert histogram ----------------
__global__ __launch_bounds__(256) void count_kernel(
        const int* __restrict__ topIdx, int* __restrict__ cnt) {
    __shared__ int hist[16];
    int tid = threadIdx.x;
    if (tid < 16) hist[tid] = 0;
    __syncthreads();
    int slot = blockIdx.x * 256 + tid;
    atomicAdd(&hist[topIdx[slot]], 1);
    __syncthreads();
    if (tid < 16) cnt[blockIdx.x * 16 + tid] = hist[tid];
}

// ---------------- K3: deterministic scan ----------------
__global__ void scan_kernel(const int* __restrict__ cnt, int* __restrict__ cbase,
                            int* __restrict__ ebase, int* __restrict__ etot) {
    __shared__ int tot_s[16];
    __shared__ int base_s[16];
    int t = threadIdx.x;
    if (t < 16) {
        int run = 0;
        for (int c = 0; c < 128; ++c) { cbase[c * 16 + t] = run; run += cnt[c * 16 + t]; }
        tot_s[t] = run;
    }
    __syncthreads();
    if (t == 0) {
        int b = 0;
        for (int e = 0; e < 16; ++e) { base_s[e] = b; b += tot_s[e]; }
    }
    __syncthreads();
    if (t < 16) {
        for (int c = 0; c < 128; ++c) cbase[c * 16 + t] += base_s[t];
        ebase[t] = base_s[t];
        etot[t]  = tot_s[t];
    }
}

// ---------------- K4: deterministic scatter ----------------
__global__ __launch_bounds__(256) void scatter_kernel(
        const int* __restrict__ topIdx, const float* __restrict__ slotW,
        const int* __restrict__ cbase,
        int* __restrict__ permTok, float* __restrict__ permW) {
    __shared__ int wcnt[8][16];
    int tid = threadIdx.x;
    int wave = tid >> 5, lane = tid & 31;
    if (tid < 128) ((int*)wcnt)[tid] = 0;
    __syncthreads();
    int slot = blockIdx.x * 256 + tid;
    int e = topIdx[slot];
    int rank = 0, tot = 0;
    for (int src = 0; src < 32; ++src) {
        int oe = __shfl(e, src);
        if (oe == e) { tot++; if (src < lane) rank++; }
    }
    if (rank == 0) wcnt[wave][e] = tot;
    __syncthreads();
    int pre = 0;
    for (int w = 0; w < wave; ++w) pre += wcnt[w][e];
    int pos = cbase[blockIdx.x * 16 + e] + pre + rank;
    permTok[pos] = slot >> 1;
    permW[pos]   = slotW[slot];
}

// ---------------- K5: f32 -> bf16 weight swizzle into B-fragment-ready layout ----------
// w1s: [e][ks(32)][nt(8)][half(2)][lane(32)][8]
// w2s: [e][nc(16)][ks(4)][nt(4)][half(2)][lane(32)][8]
// B-frag (32x16 bf16): lane = h*16 + n%16 ; element eidx -> K_local = 16h + eidx
__global__ __launch_bounds__(256) void swizzle_kernel(
        const float* __restrict__ w1, const float* __restrict__ w2,
        unsigned short* __restrict__ w1s, unsigned short* __restrict__ w2s) {
    int T = blockIdx.x * 256 + threadIdx.x;   // 0..524287
    if (T < 262144) {
        int e = T >> 14;
        int r = T & 16383;
        int ks = r >> 9;
        int r2 = r & 511;
        int nt = r2 >> 6;
        int r3 = r2 & 63;
        int half = r3 >> 5;
        int lane = r3 & 31;
        int nn = lane & 15, h = lane >> 4;
        const float* src = w1 + (size_t)e * DIM * HID;
        unsigned short* dst = w1s + (size_t)T * 8;
#pragma unroll
        for (int p = 0; p < 8; ++p) {
            int eidx = half * 8 + p;
            int K = ks * 32 + 16 * h + eidx;
            int n = nt * 16 + nn;
            dst[p] = f2bf(src[(size_t)K * HID + n]);
        }
    } else {
        int T2 = T - 262144;
        int e = T2 >> 14;
        int r = T2 & 16383;
        int nc = r >> 10;
        int r2 = r & 1023;
        int ks = r2 >> 8;
        int r3 = r2 & 255;
        int nt = r3 >> 6;
        int r4 = r3 & 63;
        int half = r4 >> 5;
        int lane = r4 & 31;
        int nn = lane & 15, h = lane >> 4;
        const float* src = w2 + (size_t)e * HID * OUTD;
        unsigned short* dst = w2s + (size_t)T2 * 8;
#pragma unroll
        for (int p = 0; p < 8; ++p) {
            int eidx = half * 8 + p;
            int K = ks * 32 + 16 * h + eidx;
            int n = nc * 64 + nt * 16 + nn;
            dst[p] = f2bf(src[(size_t)K * OUTD + n]);
        }
    }
}

// ---------------- K6: routed expert GEMMs, bf16 WMMA + async double-buffered staging ----
__global__ __launch_bounds__(128) void expert_kernel(
        const float* __restrict__ x,
        const float* __restrict__ b1, const float* __restrict__ b2,
        const unsigned short* __restrict__ w1s, const unsigned short* __restrict__ w2s,
        const int* __restrict__ permTok, const float* __restrict__ permW,
        const int* __restrict__ ebase, const int* __restrict__ etot,
        float* __restrict__ out) {
    __shared__ __align__(16) unsigned short lds_w[2][8192];   // weight double buffer, 2x16 KB
    __shared__ __align__(16) unsigned short lds_a2[2][2048];  // X A-frag double buffer, 2x4 KB
    __shared__ __align__(16) unsigned short lds_h[4096];      // H A-frags, 8 KB
    __shared__ int   lds_tok[32];
    __shared__ float lds_wt[32];

    int e = blockIdx.x;
    int tile = blockIdx.y;
    int tot = etot[e];
    if (tile * 32 >= tot) return;
    int base = ebase[e];
    int tid = threadIdx.x;
    if (tid < 32) {
        int r = tile * 32 + tid;
        if (r < tot) { lds_tok[tid] = permTok[base + r]; lds_wt[tid] = permW[base + r]; }
        else         { lds_tok[tid] = permTok[base];     lds_wt[tid] = 0.0f; }
    }
    __syncthreads();

    int wave = tid >> 5, lane = tid & 31;
    int mt = wave >> 1, npair = wave & 1;
    int nn = lane & 15, hD = lane >> 4;

    float b1v[4];
#pragma unroll
    for (int j = 0; j < 4; ++j) b1v[j] = b1[e * HID + (npair * 4 + j) * 16 + nn];

    int rtok[8]; float rwt[8];
#pragma unroll
    for (int jj = 0; jj < 8; ++jj) {
        int row = mt * 16 + jj + 8 * hD;   // D-frag: VGPR jj -> M = jj + 8*h
        rtok[jj] = lds_tok[row];
        rwt[jj]  = lds_wt[row];
    }

    // x gather assignment: 4 threads per row, 16 floats each
    int xr_row = tid >> 2, xq = tid & 3;
    const float* xrow = x + (size_t)lds_tok[xr_row] * DIM;
    int Mloc = xr_row & 15, mtl = xr_row >> 4;

    v8f acc[4];
#pragma unroll
    for (int j = 0; j < 4; ++j) acc[j] = zero8();

    float xr[16];
#define LOAD_X16(kc)                                                        \
    {                                                                       \
        const float4* xp = (const float4*)(xrow + (kc) + xq * 16);          \
        _Pragma("unroll")                                                   \
        for (int v4 = 0; v4 < 4; ++v4) {                                    \
            float4 vv = xp[v4];                                             \
            xr[v4*4+0] = vv.x; xr[v4*4+1] = vv.y;                           \
            xr[v4*4+2] = vv.z; xr[v4*4+3] = vv.w;                           \
        }                                                                   \
    }
#define STORE_X16(abuf)                                                     \
    {                                                                       \
        _Pragma("unroll")                                                   \
        for (int ii = 0; ii < 16; ++ii) {                                   \
            int kl = xq * 16 + ii;                                          \
            int s = kl >> 5, kk = kl & 31;                                  \
            int h2 = kk >> 4;                                               \
            int eix = kk & 15;                                              \
            (abuf)[(((mtl * 2 + s) * 2 + (eix >> 3)) * 32 +                 \
                    (h2 * 16 + Mloc)) * 8 + (eix & 7)] = f2bf(xr[ii]);      \
        }                                                                   \
    }

    // -------- Stage A: H = relu(X[32,1024] @ W1[1024,128] + b1) --------
    // prologue: stage chunk 0
    LOAD_X16(0);
    STORE_X16(lds_a2[0]);
    async_copy_chunk(w1s + (size_t)(e * 32) * 4096,
                     (unsigned int)(size_t)&lds_w[0][0], tid);
    wait_async0();
    __syncthreads();

    for (int ci = 0; ci < 16; ++ci) {
        int cur = ci & 1, nxt = cur ^ 1;
        if (ci < 15) {
            async_copy_chunk(w1s + (size_t)(e * 32 + (ci + 1) * 2) * 4096,
                             (unsigned int)(size_t)&lds_w[nxt][0], tid);
            LOAD_X16((ci + 1) * 64);
        }
#pragma unroll
        for (int sL = 0; sL < 2; ++sL) {
            v16bf a = load_frag(&lds_a2[cur][(mt * 2 + sL) * 512], lane);
#pragma unroll
            for (int j = 0; j < 4; ++j) {
                int nt = npair * 4 + j;
                v16bf b = load_frag(&lds_w[cur][(sL * 8 + nt) * 512], lane);
                acc[j] = __builtin_amdgcn_wmma_f32_16x16x32_bf16(
                    false, a, false, b, (short)0, acc[j], false, false);
            }
        }
        if (ci < 15) STORE_X16(lds_a2[nxt]);
        wait_async0();
        __syncthreads();
    }

    // epilogue A: bias + relu, write H into A-frag layout for stage B
#pragma unroll
    for (int j = 0; j < 4; ++j) {
        int nt = npair * 4 + j;
        int kcol = nt * 16 + nn;          // H column becomes K of stage B
        int s4 = kcol >> 5, kk = kcol & 31;
        int h2 = kk >> 4;
        int eix = kk & 15;
#pragma unroll
        for (int jj = 0; jj < 8; ++jj) {
            int Ml = jj + 8 * hD;
            float v = acc[j][jj] + b1v[j];
            v = v > 0.0f ? v : 0.0f;
            lds_h[(((mt * 4 + s4) * 2 + (eix >> 3)) * 32 + (h2 * 16 + Ml)) * 8 + (eix & 7)]
                = f2bf(v);
        }
    }
    // stage chunk 0 of W2 while H settles
    async_copy_chunk(w2s + (size_t)(e * 16) * 8192,
                     (unsigned int)(size_t)&lds_w[0][0], tid);
    wait_async0();
    __syncthreads();

    // -------- Stage B: OUT = H[32,128] @ W2[128,1024] + b2, weighted atomic scatter ------
    v16bf ha[4];
#pragma unroll
    for (int s4 = 0; s4 < 4; ++s4) ha[s4] = load_frag(&lds_h[(mt * 4 + s4) * 512], lane);

    for (int nc = 0; nc < 16; ++nc) {
        int cur = nc & 1, nxt = cur ^ 1;
        if (nc < 15) {
            async_copy_chunk(w2s + (size_t)(e * 16 + nc + 1) * 8192,
                             (unsigned int)(size_t)&lds_w[nxt][0], tid);
        }
#pragma unroll
        for (int j = 0; j < 2; ++j) {
            int nt = npair * 2 + j;
            v8f c = zero8();
#pragma unroll
            for (int s4 = 0; s4 < 4; ++s4) {
                v16bf b = load_frag(&lds_w[cur][(s4 * 4 + nt) * 512], lane);
                c = __builtin_amdgcn_wmma_f32_16x16x32_bf16(
                    false, ha[s4], false, b, (short)0, c, false, false);
            }
            int col = nc * 64 + nt * 16 + nn;
            float b2v = b2[e * OUTD + col];
#pragma unroll
            for (int jj = 0; jj < 8; ++jj) {
                float wt = rwt[jj];
                if (wt != 0.0f) {
                    atomicAdd(out + (size_t)rtok[jj] * OUTD + col, wt * (c[jj] + b2v));
                }
            }
        }
        wait_async0();
        __syncthreads();
    }
#undef LOAD_X16
#undef STORE_X16
}

// ---------------- K7: aux loss ----------------
__global__ void aux_kernel(const float* __restrict__ gpart, float* __restrict__ out_aux) {
    __shared__ float imp[16];
    int t = threadIdx.x;
    if (t < 16) {
        float s = 0.f;
        for (int b = 0; b < 256; ++b) s += gpart[b * 16 + t];
        imp[t] = s / (float)N_TOK;
    }
    __syncthreads();
    if (t == 0) {
        const float target = 1.0f / 16.0f;
        float aux = 0.f;
        for (int e = 0; e < 16; ++e) aux += target * (logf(target) - logf(imp[e]));
        out_aux[0] = aux;
    }
}

extern "C" void kernel_launch(void* const* d_in, const int* in_sizes, int n_in,
                              void* d_out, int out_size, void* d_ws, size_t ws_size,
                              hipStream_t stream) {
    const float* x  = (const float*)d_in[0];
    const float* wg = (const float*)d_in[1];
    const float* w1 = (const float*)d_in[2];
    const float* b1 = (const float*)d_in[3];
    const float* w2 = (const float*)d_in[4];
    const float* b2 = (const float*)d_in[5];
    float* out = (float*)d_out;

    char* ws = (char*)d_ws;
    int*   topIdx  = (int*)  (ws + 0);         // 128 KB
    float* slotW   = (float*)(ws + 131072);    // 128 KB
    float* gpart   = (float*)(ws + 262144);    // 16 KB
    int*   cnt     = (int*)  (ws + 278528);    // 8 KB
    int*   cbase   = (int*)  (ws + 286720);    // 8 KB
    int*   ebaseP  = (int*)  (ws + 294912);    // 64 B
    int*   etotP   = (int*)  (ws + 295168);    // 64 B
    int*   permTok = (int*)  (ws + 295424);    // 128 KB
    float* permW   = (float*)(ws + 426496);    // 128 KB
    unsigned short* w1s = (unsigned short*)(ws + 557568);   // 4 MB
    unsigned short* w2s = (unsigned short*)(ws + 4751872);  // 4 MB  (total ~8.95 MB)

    long long outN = (long long)N_TOK * OUTD + 1;
    zero_kernel   <<<2048, 256, 0, stream>>>(out, outN);
    gate_kernel   <<<256,  256, 0, stream>>>(x, wg, topIdx, slotW, gpart);
    count_kernel  <<<128,  256, 0, stream>>>(topIdx, cnt);
    scan_kernel   <<<1,    32,  0, stream>>>(cnt, cbase, ebaseP, etotP);
    scatter_kernel<<<128,  256, 0, stream>>>(topIdx, slotW, cbase, permTok, permW);
    swizzle_kernel<<<2048, 256, 0, stream>>>(w1, w2, w1s, w2s);
    dim3 g6(16, 512);
    expert_kernel <<<g6,   128, 0, stream>>>(x, b1, b2, w1s, w2s, permTok, permW,
                                             ebaseP, etotP, out);
    aux_kernel    <<<1,    32,  0, stream>>>(gpart, out + (size_t)N_TOK * OUTD);
}
